// PyLSTM_78486232367304
// MI455X (gfx1250) — matrix operand order repaired and checked by reading, not compiled, not run on hardware
//
#include <hip/hip_runtime.h>
#include <hip/hip_bf16.h>

// Problem dims (from reference): B=128, T=512, D=64, H=512
#define B_  128
#define T_  512
#define D_  64
#define H_  512
#define G4  (4 * H_)      // 2048 gate columns
#define K1  (D_ + H_)     // 576  (layer-1 K)
#define K2  (2 * H_)      // 1024 (layer-2 K)

typedef __attribute__((ext_vector_type(16))) __bf16 v16bf;
typedef __attribute__((ext_vector_type(8)))  __bf16 v8bf;
typedef __attribute__((ext_vector_type(8)))  float  v8f;

static __device__ __forceinline__ float sigmoidf_(float x) {
    return 1.0f / (1.0f + __expf(-x));
}

// ---- WMMA fragment loaders ---------------------------------------------
// A fragment (16x32, 16-bit): lane half 0 holds K = {kb+0..7, kb+16..23},
// half 1 holds K = {kb+8..15, kb+24..31}; element e<8 -> kb+half*8+e,
// element e>=8 -> kb+16+half*8+(e-8).  (ISA 7.12.2, 16-bit A 16x32)
static __device__ __forceinline__ v16bf load_a_f32(const float* __restrict__ row,
                                                   int kb, int half) {
    const float* p0 = row + kb + half * 8;
    const float* p1 = p0 + 16;
    v16bf a;
#pragma unroll
    for (int e = 0; e < 8; ++e) { a[e] = (__bf16)p0[e]; a[e + 8] = (__bf16)p1[e]; }
    return a;
}
static __device__ __forceinline__ v16bf load_a_bf16(const __bf16* __restrict__ row,
                                                    int kb, int half) {
    v8bf lo = *(const v8bf*)(row + kb + half * 8);
    v8bf hi = *(const v8bf*)(row + kb + half * 8 + 16);
    v16bf a;
#pragma unroll
    for (int e = 0; e < 8; ++e) { a[e] = lo[e]; a[e + 8] = hi[e]; }
    return a;
}
// B fragment (32x16, 16-bit): lane = N; same K order as A: element e<8 ->
// kb+half*8+e, element e>=8 -> kb+16+half*8+(e-8).
// Since gates = combined @ W^T, B[k][n] = W[n][k]: per-lane 2x16B (16 elems apart)
// load from W row n.
static __device__ __forceinline__ v16bf load_b_bf16(const __bf16* __restrict__ wrow,
                                                    int kb, int half) {
    v8bf lo = *(const v8bf*)(wrow + kb + half * 8);
    v8bf hi = *(const v8bf*)(wrow + kb + half * 8 + 16);
    v16bf b;
#pragma unroll
    for (int e = 0; e < 8; ++e) { b[e] = lo[e]; b[e + 8] = hi[e]; }
    return b;
}

// ---- weight f32 -> bf16 conversion (once per call; stays L2-resident) --
__global__ __launch_bounds__(256) void k_convert(const float* __restrict__ W1,
                                                 const float* __restrict__ W2,
                                                 __bf16* __restrict__ W1b,
                                                 __bf16* __restrict__ W2b) {
    int i = blockIdx.x * 256 + threadIdx.x;
    const int n1 = G4 * K1;
    const int n2 = G4 * K2;
    if (i < n1) W1b[i] = (__bf16)W1[i];
    if (i < n2) W2b[i] = (__bf16)W2[i];
}

__global__ __launch_bounds__(256) void k_init(float* c1, float* c2,
                                              float* h1f, float* h2f, float* r1f,
                                              __bf16* h1b, __bf16* h2b, __bf16* r1b) {
    int i = blockIdx.x * 256 + threadIdx.x;
    if (i < B_ * H_) {
        c1[i] = 0.f; c2[i] = 0.f; h1f[i] = 0.f; h2f[i] = 0.f; r1f[i] = 0.f;
        h1b[i] = (__bf16)0.f; h2b[i] = (__bf16)0.f; r1b[i] = (__bf16)0.f;
    }
}

// ---- layer 1: gates GEMM (bf16 WMMA) + LSTM cell + LayerNorm -----------
// grid = 8 blocks (16 batch rows each), block = 256 threads = 8 waves.
// Wave w owns j-tiles w*4..w*4+3; for each it computes i/f/g/o 16x16 tiles.
__global__ __launch_bounds__(256) void k_layer1(
    const float* __restrict__ x, int t,
    const __bf16* __restrict__ W1b,
    const float* __restrict__ b1, const float* __restrict__ g1,
    const float* __restrict__ be1,
    __bf16* __restrict__ h1b, float* __restrict__ c1,
    float* __restrict__ h1f, __bf16* __restrict__ r1b, float* __restrict__ r1f)
{
    __shared__ float sA[16 * 512];
    __shared__ float sMu[16], sRs[16];
    const int tid  = threadIdx.x;
    const int wave = tid >> 5, lane = tid & 31;
    const int half = lane >> 4, ln16 = lane & 15;
    const int mb   = blockIdx.x;
    const int bm   = mb * 16 + ln16;                 // A-matrix row (batch idx)
    const float*  xrow = x   + ((size_t)bm * T_ + t) * D_;
    const __bf16* hrow = h1b + (size_t)bm * H_;

    for (int jj = 0; jj < 4; ++jj) {
        const int j = (wave * 4 + jj) * 16 + ln16;   // hidden unit (B-lane N)
        v8f ai = {}, af = {}, ag = {}, ao = {};
        const __bf16* wi = W1b + (size_t)(j)          * K1;
        const __bf16* wf = W1b + (size_t)(j + 512)    * K1;
        const __bf16* wg = W1b + (size_t)(j + 1024)   * K1;
        const __bf16* wo = W1b + (size_t)(j + 1536)   * K1;
        for (int ks = 0; ks < K1 / 32; ++ks) {       // 18 k-steps
            const int kb = ks * 32;
            v16bf a = (ks < 2) ? load_a_f32(xrow, kb, half)        // x part (K<64)
                               : load_a_bf16(hrow, kb - 64, half); // h1 part
            v16bf bi = load_b_bf16(wi, kb, half);
            v16bf bff = load_b_bf16(wf, kb, half);
            v16bf bg = load_b_bf16(wg, kb, half);
            v16bf bo = load_b_bf16(wo, kb, half);
            ai = __builtin_amdgcn_wmma_f32_16x16x32_bf16(false, a, false, bi,  (short)0, ai, false, false);
            af = __builtin_amdgcn_wmma_f32_16x16x32_bf16(false, a, false, bff, (short)0, af, false, false);
            ag = __builtin_amdgcn_wmma_f32_16x16x32_bf16(false, a, false, bg,  (short)0, ag, false, false);
            ao = __builtin_amdgcn_wmma_f32_16x16x32_bf16(false, a, false, bo,  (short)0, ao, false, false);
        }
        // C/D layout: element e -> local row (e + 8*half), col = ln16.
        // i/f/g/o accumulators align element-wise -> register-resident cell.
        const float bbi = b1[j], bbf = b1[j + 512], bbg = b1[j + 1024], bbo = b1[j + 1536];
#pragma unroll
        for (int e = 0; e < 8; ++e) {
            const int rl  = e + half * 8;
            const int bmr = mb * 16 + rl;
            float iv = sigmoidf_(ai[e] + bbi);
            float fv = sigmoidf_(af[e] + bbf);
            float gv = tanhf(ag[e] + bbg);
            float ov = sigmoidf_(ao[e] + bbo);
            size_t ci = (size_t)bmr * H_ + j;
            float cn = fv * c1[ci] + iv * gv;
            c1[ci] = cn;
            sA[rl * 512 + j] = ov * tanhf(cn);
        }
    }
    __syncthreads();
    // LayerNorm stats: each wave reduces 2 rows across 512 cols (wave32 shfl)
#pragma unroll
    for (int rr = 0; rr < 2; ++rr) {
        int r = wave * 2 + rr;
        float s = 0.f, sq = 0.f;
        for (int jc = lane; jc < 512; jc += 32) {
            float v = sA[r * 512 + jc]; s += v; sq += v * v;
        }
#pragma unroll
        for (int off = 16; off > 0; off >>= 1) {
            s  += __shfl_down(s,  off, 32);
            sq += __shfl_down(sq, off, 32);
        }
        if (lane == 0) {
            float mu  = s  * (1.f / 512.f);
            float var = sq * (1.f / 512.f) - mu * mu;
            sMu[r] = mu; sRs[r] = rsqrtf(var + 1e-5f);
        }
    }
    __syncthreads();
    for (int idx = tid; idx < 16 * 512; idx += 256) {
        int r = idx >> 9, jc = idx & 511;
        int bmr = mb * 16 + r;
        float h = (sA[idx] - sMu[r]) * sRs[r] * g1[jc] + be1[jc];
        size_t o = (size_t)bmr * H_ + jc;
        h1b[o] = (__bf16)h;          // recurrent input for layer-1 next t
        h1f[o] = h;                  // final-state output
        float hr = fmaxf(h, 0.f);
        r1b[o] = (__bf16)hr;         // feeds layer-2 GEMM
        r1f[o] = hr;                 // residual (f32 fidelity)
    }
}

// ---- layer 2 + residual + scalar head ----------------------------------
__global__ __launch_bounds__(256) void k_layer2(
    int t,
    const __bf16* __restrict__ W2b,
    const float* __restrict__ b2, const float* __restrict__ g2,
    const float* __restrict__ be2,
    const __bf16* __restrict__ r1b, const float* __restrict__ r1f,
    __bf16* __restrict__ h2b, float* __restrict__ c2, float* __restrict__ h2f,
    const float* __restrict__ Wf, const float* __restrict__ bf0,
    float* __restrict__ out)
{
    __shared__ float sA[16 * 512];
    __shared__ float sMu[16], sRs[16];
    const int tid  = threadIdx.x;
    const int wave = tid >> 5, lane = tid & 31;
    const int half = lane >> 4, ln16 = lane & 15;
    const int mb   = blockIdx.x;
    const int bm   = mb * 16 + ln16;
    const __bf16* rrow = r1b + (size_t)bm * H_;
    const __bf16* hrow = h2b + (size_t)bm * H_;

    for (int jj = 0; jj < 4; ++jj) {
        const int j = (wave * 4 + jj) * 16 + ln16;
        v8f ai = {}, af = {}, ag = {}, ao = {};
        const __bf16* wi = W2b + (size_t)(j)          * K2;
        const __bf16* wf = W2b + (size_t)(j + 512)    * K2;
        const __bf16* wg = W2b + (size_t)(j + 1024)   * K2;
        const __bf16* wo = W2b + (size_t)(j + 1536)   * K2;
        for (int ks = 0; ks < K2 / 32; ++ks) {       // 32 k-steps
            const int kb = ks * 32;
            v16bf a = (ks < 16) ? load_a_bf16(rrow, kb, half)        // relu(h1)
                                : load_a_bf16(hrow, kb - 512, half); // h2_prev
            v16bf bi = load_b_bf16(wi, kb, half);
            v16bf bff = load_b_bf16(wf, kb, half);
            v16bf bg = load_b_bf16(wg, kb, half);
            v16bf bo = load_b_bf16(wo, kb, half);
            ai = __builtin_amdgcn_wmma_f32_16x16x32_bf16(false, a, false, bi,  (short)0, ai, false, false);
            af = __builtin_amdgcn_wmma_f32_16x16x32_bf16(false, a, false, bff, (short)0, af, false, false);
            ag = __builtin_amdgcn_wmma_f32_16x16x32_bf16(false, a, false, bg,  (short)0, ag, false, false);
            ao = __builtin_amdgcn_wmma_f32_16x16x32_bf16(false, a, false, bo,  (short)0, ao, false, false);
        }
        const float bbi = b2[j], bbf = b2[j + 512], bbg = b2[j + 1024], bbo = b2[j + 1536];
#pragma unroll
        for (int e = 0; e < 8; ++e) {
            const int rl  = e + half * 8;
            const int bmr = mb * 16 + rl;
            float iv = sigmoidf_(ai[e] + bbi);
            float fv = sigmoidf_(af[e] + bbf);
            float gv = tanhf(ag[e] + bbg);
            float ov = sigmoidf_(ao[e] + bbo);
            size_t ci = (size_t)bmr * H_ + j;
            float cn = fv * c2[ci] + iv * gv;
            c2[ci] = cn;
            sA[rl * 512 + j] = ov * tanhf(cn);
        }
    }
    __syncthreads();
#pragma unroll
    for (int rr = 0; rr < 2; ++rr) {
        int r = wave * 2 + rr;
        float s = 0.f, sq = 0.f;
        for (int jc = lane; jc < 512; jc += 32) {
            float v = sA[r * 512 + jc]; s += v; sq += v * v;
        }
#pragma unroll
        for (int off = 16; off > 0; off >>= 1) {
            s  += __shfl_down(s,  off, 32);
            sq += __shfl_down(sq, off, 32);
        }
        if (lane == 0) {
            float mu  = s  * (1.f / 512.f);
            float var = sq * (1.f / 512.f) - mu * mu;
            sMu[r] = mu; sRs[r] = rsqrtf(var + 1e-5f);
        }
    }
    __syncthreads();
    // normalize, residual mix, stash per-element head contributions in LDS
    for (int idx = tid; idx < 16 * 512; idx += 256) {
        int r = idx >> 9, jc = idx & 511;
        int bmr = mb * 16 + r;
        float h = (sA[idx] - sMu[r]) * sRs[r] * g2[jc] + be2[jc];
        size_t o = (size_t)bmr * H_ + jc;
        h2b[o] = (__bf16)h;
        h2f[o] = h;
        float hr = fmaxf(h, 0.f);
        float hm = (hr + 0.5f * r1f[o]) * (1.f / 1.5f);
        sA[idx] = hm * Wf[jc];       // head dot-product contribution
    }
    __syncthreads();
#pragma unroll
    for (int rr = 0; rr < 2; ++rr) {
        int r = wave * 2 + rr;
        float s = 0.f;
        for (int jc = lane; jc < 512; jc += 32) s += sA[r * 512 + jc];
#pragma unroll
        for (int off = 16; off > 0; off >>= 1) s += __shfl_down(s, off, 32);
        if (lane == 0) {
            float ov = s + bf0[0];
            size_t p = (size_t)(mb * 16 + r) * T_ + t;
            out[p] = ov;                       // outputs (first copy)
            out[(size_t)B_ * T_ + p] = ov;     // outputs (second copy, tuple)
        }
    }
}

__global__ __launch_bounds__(256) void k_finalize(const float* __restrict__ h1f,
                                                  const float* __restrict__ c1,
                                                  const float* __restrict__ h2f,
                                                  const float* __restrict__ c2,
                                                  float* __restrict__ out) {
    int i = blockIdx.x * 256 + threadIdx.x;
    const int N = B_ * H_;                      // 65536
    const size_t base = (size_t)2 * B_ * T_;    // after two output copies
    if (i < N) {
        out[base + 0 * N + i] = h1f[i];
        out[base + 1 * N + i] = c1[i];
        out[base + 2 * N + i] = h2f[i];
        out[base + 3 * N + i] = c2[i];
    }
}

extern "C" void kernel_launch(void* const* d_in, const int* in_sizes, int n_in,
                              void* d_out, int out_size, void* d_ws, size_t ws_size,
                              hipStream_t stream) {
    (void)in_sizes; (void)n_in; (void)out_size; (void)ws_size;
    const float* x   = (const float*)d_in[0];
    const float* W1  = (const float*)d_in[1];
    const float* b1  = (const float*)d_in[2];
    const float* g1  = (const float*)d_in[3];
    const float* be1 = (const float*)d_in[4];
    const float* W2  = (const float*)d_in[5];
    const float* b2  = (const float*)d_in[6];
    const float* g2  = (const float*)d_in[7];
    const float* be2 = (const float*)d_in[8];
    const float* Wf  = (const float*)d_in[9];
    const float* bf0 = (const float*)d_in[10];
    float* out = (float*)d_out;

    char* ws = (char*)d_ws;
    size_t off = 0;
    auto alloc = [&](size_t bytes) -> char* {
        char* p = ws + off;
        off += (bytes + 255) & ~(size_t)255;
        return p;
    };
    __bf16* W1b = (__bf16*)alloc((size_t)G4 * K1 * 2);   // 2.25 MB
    __bf16* W2b = (__bf16*)alloc((size_t)G4 * K2 * 2);   // 4.0 MB
    __bf16* h1b = (__bf16*)alloc((size_t)B_ * H_ * 2);
    __bf16* h2b = (__bf16*)alloc((size_t)B_ * H_ * 2);
    __bf16* r1b = (__bf16*)alloc((size_t)B_ * H_ * 2);
    float* c1  = (float*)alloc((size_t)B_ * H_ * 4);
    float* c2  = (float*)alloc((size_t)B_ * H_ * 4);
    float* h1f = (float*)alloc((size_t)B_ * H_ * 4);
    float* h2f = (float*)alloc((size_t)B_ * H_ * 4);
    float* r1f = (float*)alloc((size_t)B_ * H_ * 4);

    k_convert<<<(G4 * K2 + 255) / 256, 256, 0, stream>>>(W1, W2, W1b, W2b);
    k_init<<<(B_ * H_ + 255) / 256, 256, 0, stream>>>(c1, c2, h1f, h2f, r1f,
                                                      h1b, h2b, r1b);
    for (int t = 0; t < T_; ++t) {
        k_layer1<<<B_ / 16, 256, 0, stream>>>(x, t, W1b, b1, g1, be1,
                                              h1b, c1, h1f, r1b, r1f);
        k_layer2<<<B_ / 16, 256, 0, stream>>>(t, W2b, b2, g2, be2,
                                              r1b, r1f, h2b, c2, h2f,
                                              Wf, bf0, out);
    }
    k_finalize<<<(B_ * H_ + 255) / 256, 256, 0, stream>>>(h1f, c1, h2f, c2, out);
}
